// FNO3d_52063593562981
// MI455X (gfx1250) — matrix operand order repaired and checked by reading, not compiled, not run on hardware
//
#include <hip/hip_runtime.h>
#include <hip/hip_bf16.h>
#include <math.h>

typedef __attribute__((ext_vector_type(16))) _Float16 v16h;
typedef __attribute__((ext_vector_type(4)))  _Float16 v4h;
typedef __attribute__((ext_vector_type(8)))  float    v8f;
typedef __attribute__((ext_vector_type(4)))  float    v4f;

#define NN 64
#define CC 32
#define HH 128
#define WW 128
#define PLANE (HH*WW)          // 16384
#define NROWS (NN*CC*HH)       // 262144 rows of length 128
#define NPTS  (NN*PLANE)       // 1048576 spatial points
#define PI_F 3.14159265358979323846f

__device__ __forceinline__ float gelu_exact(float x) {
  return 0.5f * x * (1.0f + erff(x * 0.7071067811865476f));
}

// ---------------------------------------------------------------------------
// WMMA fragment helpers (CDNA5 16-bit layouts, wave32)
// A fragment element order: a[2j],a[2j+1] = K = k0(j), k0(j)+1 where
//   k0(j) = (j<4) ? 2j + 8*hi : 16 + 2(j-4) + 8*hi,  row M = lane&15, hi = lane>>4
// B fragment: lane = hi*16+col, N = col; K order as in paired A (table; _chan A, skF, fc1F: K = hi*16 + j)
// C fragment: VGPR r -> M = r + 8*hi, N = lane&15
// ---------------------------------------------------------------------------

// B fragment from a table pre-packed in fragment order: 32 lanes x 16 halfs.
__device__ __forceinline__ v16h load_frag_packed(const _Float16* base) {
  int lane = threadIdx.x & 31;
  return *(const v16h*)(base + lane * 16);   // 32B contiguous per lane
}

// A fragment from row-major f32 global rows (stride ldr floats), K window kbase..kbase+31.
__device__ __forceinline__ v16h load_frag_a_rows(const float* rowp, int ldr, int kbase) {
  int lane = threadIdx.x & 31;
  int row = lane & 15, hi = lane >> 4;
  const float* p = rowp + (size_t)row * ldr + kbase + hi * 8;
  v4f f0 = *(const v4f*)(p);
  v4f f1 = *(const v4f*)(p + 4);
  v4f f2 = *(const v4f*)(p + 16);
  v4f f3 = *(const v4f*)(p + 20);
  v16h a;
#pragma unroll
  for (int i = 0; i < 4; ++i) {
    a[i]      = (_Float16)f0[i];
    a[4 + i]  = (_Float16)f1[i];
    a[8 + i]  = (_Float16)f2[i];
    a[12 + i] = (_Float16)f3[i];
  }
  return a;
}

// A fragment with K=16 real columns, zero-padded to 32 (for the inverse-W DFT).
__device__ __forceinline__ v16h load_frag_a_pad16(const float* rowp, int ldr) {
  int lane = threadIdx.x & 31;
  int row = lane & 15, hi = lane >> 4;
  const float* p = rowp + (size_t)row * ldr + hi * 8;
  v4f f0 = *(const v4f*)(p);
  v4f f1 = *(const v4f*)(p + 4);
  v16h a = {};
#pragma unroll
  for (int i = 0; i < 4; ++i) {
    a[i]     = (_Float16)f0[i];
    a[4 + i] = (_Float16)f1[i];
  }
  return a;
}

// A fragment where K = channel (stride PLANE floats), M = 16 consecutive points.
// Coalesced across lanes (lanes 0..15 read consecutive points).
__device__ __forceinline__ v16h load_frag_a_chan(const float* hp) {
  int lane = threadIdx.x & 31;
  int pt = lane & 15, hi = lane >> 4;
  v16h a = {};
#pragma unroll
  for (int j = 0; j < 8; ++j) {
    int c0 = 2*j + hi*16;
    a[2*j]   = (_Float16)hp[(size_t)c0 * PLANE + pt];
    a[2*j+1] = (_Float16)hp[(size_t)(c0+1) * PLANE + pt];
  }
  return a;
}

__device__ __forceinline__ void store_frag_c(float* out, int ldo, v8f c) {
  int lane = threadIdx.x & 31;
  int col = lane & 15, hi = lane >> 4;
#pragma unroll
  for (int r = 0; r < 8; ++r)
    out[(size_t)(r + 8*hi)*ldo + col] = c[r];
}

// ---------------------------------------------------------------------------
// Tables (fragment-order packed twiddles) + per-call weight packing
// EwF  [4 ktile][32 lane][16] : forward W-DFT basis, B tile for K rows kt*32..+31
// EwiF [8 ntile][32 lane][16] : inverse W-DFT basis (1/HW + hermitian 2x), K rows>=16 zero
// E1   [16][128][2] f32 : e^{-2pi i k1 y/128}, k1 in {0..7,120..127}
// E1i  [128][16][2] f32 : e^{+2pi i k1 y/128}
// ---------------------------------------------------------------------------
__global__ void fno_init_tables(_Float16* __restrict__ EwF, _Float16* __restrict__ EwiF,
                                float* __restrict__ E1, float* __restrict__ E1i) {
  int i = blockIdx.x * blockDim.x + threadIdx.x;
  if (i < 4*512) {                       // EwF
    int kt = i >> 9, r = i & 511;
    int lane = r >> 4, j = r & 15;
    int col = lane & 15;
    int krow = (j < 8) ? (((lane >> 4) << 3) + j) : (16 + ((lane >> 4) << 3) + j - 8);   // K within tile: table order
    int x = kt*32 + krow;                // spatial index
    int k2 = col >> 1;
    float th = 2.f*PI_F * (float)(k2 * x) / 128.f;
    EwF[i] = (_Float16)((col & 1) ? -sinf(th) : cosf(th));
  }
  if (i < 8*512) {                       // EwiF
    int nt = i >> 9, r = i & 511;
    int lane = r >> 4, j = r & 15;
    int x = nt*16 + (lane & 15);
    int krow = (j < 8) ? (((lane >> 4) << 3) + j) : (16 + ((lane >> 4) << 3) + j - 8);
    float v = 0.f;
    if (krow < 16) {
      int k = krow >> 1;
      float coef = ((k == 0) ? 1.f : 2.f) / 16384.f;
      float th = 2.f*PI_F * (float)(k * x) / 128.f;
      v = (krow & 1) ? (-coef * sinf(th)) : (coef * cosf(th));
    }
    EwiF[i] = (_Float16)v;
  }
  if (i < 16*128) {                      // E1
    int j1 = i >> 7, y = i & 127;
    int k1 = (j1 < 8) ? j1 : (112 + j1);
    float ph = 2.f*PI_F * (float)(k1*y) / 128.f;
    E1[2*i]   = cosf(ph);
    E1[2*i+1] = -sinf(ph);
  }
  if (i < 128*16) {                      // E1i
    int y = i >> 4, j1 = i & 15;
    int k1 = (j1 < 8) ? j1 : (112 + j1);
    float ph = 2.f*PI_F * (float)(k1*y) / 128.f;
    E1i[2*i]   = cosf(ph);
    E1i[2*i+1] = sinf(ph);
  }
}

// skF  [4 layer][2 otile][32][16] from skip_w[L][o][c]  (B[K=c][N=o])
// fc1F [8 ntile][32][16]          from fc1_w[c][j]      (B[K=c][N=j])
__global__ void fno_pack_weights(const float* __restrict__ skw, const float* __restrict__ fc1w,
                                 _Float16* __restrict__ skF, _Float16* __restrict__ fc1F) {
  int i = blockIdx.x * blockDim.x + threadIdx.x;
  if (i < 4096) {                        // skF
    int layer = i >> 10, r2 = i & 1023;
    int ot = r2 >> 9, r = r2 & 511;
    int lane = r >> 4, j = r & 15;
    int o = ot*16 + (lane & 15);
    int cin = ((lane >> 4) << 4) + j;
    skF[i] = (_Float16)skw[layer*1024 + o*32 + cin];
  }
  if (i < 4096) {                        // fc1F
    int nt = i >> 9, r = i & 511;
    int lane = r >> 4, j = r & 15;
    int colj = nt*16 + (lane & 15);
    int cin = ((lane >> 4) << 4) + j;
    fc1F[i] = (_Float16)fc1w[cin*128 + colj];
  }
}

// ---- fc0: [NPTS,4] @ [4,32] + b, write channel-major [N,C,H,W] ----
__global__ void fno_fc0(const float* __restrict__ xin, const float* __restrict__ w,
                        const float* __restrict__ b, float* __restrict__ h0) {
  int p = blockIdx.x * blockDim.x + threadIdx.x;
  if (p >= NPTS) return;
  int n = p >> 14, pos = p & (PLANE-1);
  int y = pos >> 7, x = pos & 127;
  int bb = n >> 4, t = n & 15;
  const float* xp = xin + (((size_t)((bb*HH + y)*WW + x) * 16 + t) * 4);
  float d0 = xp[0], d1 = xp[1], d2 = xp[2], d3 = xp[3];
#pragma unroll
  for (int c = 0; c < CC; ++c) {
    float acc = b[c] + d0*w[c] + d1*w[32+c] + d2*w[64+c] + d3*w[96+c];
    h0[(size_t)(n*CC + c) * PLANE + pos] = acc;
  }
}

// ---- forward partial DFT along W: [NROWS,128] @ Ew[128,16] -> U[NROWS,16] ----
__global__ void __launch_bounds__(256)
fno_dft_w(const float* __restrict__ h, float* __restrict__ U,
          const _Float16* __restrict__ EwF) {
  int wave = threadIdx.x >> 5;
  int rowbase = (blockIdx.x * 8 + wave) * 16;
  const float* rowp = h + (size_t)rowbase * 128;
  __builtin_prefetch(rowp, 0, 1);        // global_prefetch_b8
  v8f c = {};
#pragma unroll
  for (int kt = 0; kt < 4; ++kt) {
    v16h a  = load_frag_a_rows(rowp, 128, kt*32);
    v16h bf = load_frag_packed(EwF + kt*512);
    c = __builtin_amdgcn_wmma_f32_16x16x32_f16(false, a, false, bf, (short)0, c, false, false);
  }
  store_frag_c(U + (size_t)rowbase * 16, 16, c);
}

// ---- forward DFT along H on 16 kept k1 modes (VALU, small) ----
__global__ void fno_dft_h(const float* __restrict__ U, float* __restrict__ V,
                          const float* __restrict__ E1) {
  int i = blockIdx.x * blockDim.x + threadIdx.x;   // 262144 = (n,c)*16*8
  int k2 = i & 7, j1 = (i >> 3) & 15, nc = i >> 7;
  const float* up = U + (size_t)nc * (128*16) + 2*k2;
  const float* ep = E1 + j1 * 256;
  float vr = 0.f, vi = 0.f;
  for (int y = 0; y < 128; ++y) {
    float ur = up[y*16], ui = up[y*16+1];
    float cc = ep[2*y], ss = ep[2*y+1];
    vr += ur*cc - ui*ss;
    vi += ur*ss + ui*cc;
  }
  float* vp = V + ((size_t)nc * 16 + j1) * 16 + 2*k2;
  vp[0] = vr; vp[1] = vi;
}

// ---- spectral channel mix: out[o] = sum_i V[i] * W[i,o] (complex) ----
__global__ void fno_specmul(const float* __restrict__ V, float* __restrict__ Vo,
                            const float* __restrict__ w1r, const float* __restrict__ w1i,
                            const float* __restrict__ w2r, const float* __restrict__ w2i) {
  int i = blockIdx.x * blockDim.x + threadIdx.x;   // 262144 = n*16*8*32
  int o = i & 31, k2 = (i >> 5) & 7, j1 = (i >> 8) & 15, n = i >> 12;
  const float *wr, *wi;
  int jm = j1 & 7;
  if (j1 < 8) { wr = w1r; wi = w1i; } else { wr = w2r; wi = w2i; }
  int woff = (o * 8 + jm) * 8 + k2;                // w[ci][o][jm][k2], ci stride 2048
  float ar = 0.f, ai = 0.f;
  for (int ci = 0; ci < 32; ++ci) {
    const float* vp = V + (((size_t)(n*32 + ci) * 16 + j1) * 16) + 2*k2;
    float vr = vp[0], vi_ = vp[1];
    float wrr = wr[ci*2048 + woff], wii = wi[ci*2048 + woff];
    ar += vr*wrr - vi_*wii;
    ai += vr*wii + vi_*wrr;
  }
  float* op = Vo + (((size_t)(n*32 + o) * 16 + j1) * 16) + 2*k2;
  op[0] = ar; op[1] = ai;
}

// ---- inverse DFT along H from 16 modes -> Z[NROWS,16] ----
__global__ void fno_idft_h(const float* __restrict__ Vo, float* __restrict__ Z,
                           const float* __restrict__ E1i) {
  int i = blockIdx.x * blockDim.x + threadIdx.x;   // 2097152 = (n,c)*128*8
  int k2 = i & 7, y = (i >> 3) & 127, nc = i >> 10;
  const float* ep = E1i + y * 32;
  const float* vp = Vo + (size_t)nc * 256 + 2*k2;
  float zr = 0.f, zi = 0.f;
#pragma unroll
  for (int j1 = 0; j1 < 16; ++j1) {
    float vr = vp[j1*16], vi = vp[j1*16+1];
    float cc = ep[2*j1], ss = ep[2*j1+1];
    zr += vr*cc - vi*ss;
    zi += vr*ss + vi*cc;
  }
  float* zp = Z + ((size_t)nc * 128 + y) * 16 + 2*k2;
  zp[0] = zr; zp[1] = zi;
}

// ---- inverse rDFT along W: [NROWS,16(pad32)] @ Ewi[32,128] -> X1[NROWS,128] ----
__global__ void __launch_bounds__(256)
fno_idft_w(const float* __restrict__ Z, float* __restrict__ X1,
           const _Float16* __restrict__ EwiF) {
  int wave = threadIdx.x >> 5;
  int rowbase = (blockIdx.x * 8 + wave) * 16;
  v16h a = load_frag_a_pad16(Z + (size_t)rowbase * 16, 16);
  float* xp = X1 + (size_t)rowbase * 128;
#pragma unroll
  for (int nt = 0; nt < 8; ++nt) {
    v16h bf = load_frag_packed(EwiF + nt*512);
    v8f c = {};
    c = __builtin_amdgcn_wmma_f32_16x16x32_f16(false, a, false, bf, (short)0, c, false, false);
    store_frag_c(xp + nt*16, 128, c);
  }
}

// ---- skip 1x1 conv + bias + add spectral + (gelu), in-place into X1 buffer ----
__global__ void __launch_bounds__(256)
fno_skip(const float* __restrict__ h, float* __restrict__ X1io,
         const _Float16* __restrict__ skF, const float* __restrict__ skb, int do_gelu) {
  int lane = threadIdx.x & 31, wave = threadIdx.x >> 5;
  int pbase = (blockIdx.x * 8 + wave) * 16;
  int n = pbase >> 14, pos = pbase & (PLANE-1);
  const float* hp = h + (size_t)n * CC * PLANE + pos;
  v16h a = load_frag_a_chan(hp);
  int col = lane & 15, hi = lane >> 4;
  float* xbase = X1io + (size_t)n * CC * PLANE + pos;
#pragma unroll
  for (int ot = 0; ot < 2; ++ot) {
    v16h bf = load_frag_packed(skF + ot*512);
    v8f c = {};
    c = __builtin_amdgcn_wmma_f32_16x16x32_f16(false, a, false, bf, (short)0, c, false, false);
    int o = ot*16 + col;
    float bias = skb[o];
    float* xo = xbase + (size_t)o * PLANE;
#pragma unroll
    for (int r = 0; r < 8; ++r) {
      int pt = r + 8*hi;
      float v = xo[pt] + c[r] + bias;
      if (do_gelu) v = gelu_exact(v);
      xo[pt] = v;    // each lane reads then writes the same address: in-place safe
    }
  }
}

// ---- fused fc1 + gelu + fc2 -> output [B,H,W,T,1] ----
__global__ void __launch_bounds__(256)
fno_fc(const float* __restrict__ h, const _Float16* __restrict__ fc1F,
       const float* __restrict__ fc1b, const float* __restrict__ fc2w,
       const float* __restrict__ fc2b, float* __restrict__ out) {
  __shared__ float sHid[8][16*128];      // 64 KB: per-wave hidden tile
  int lane = threadIdx.x & 31, wave = threadIdx.x >> 5;
  int pbase = (blockIdx.x * 8 + wave) * 16;
  int n = pbase >> 14, pos = pbase & (PLANE-1);
  const float* hp = h + (size_t)n * CC * PLANE + pos;
  v16h a = load_frag_a_chan(hp);
  int col = lane & 15, hi = lane >> 4;
#pragma unroll
  for (int nt = 0; nt < 8; ++nt) {
    v16h bf = load_frag_packed(fc1F + nt*512);
    v8f c = {};
    c = __builtin_amdgcn_wmma_f32_16x16x32_f16(false, a, false, bf, (short)0, c, false, false);
    int j = nt*16 + col;
    float bias = fc1b[j];
#pragma unroll
    for (int r = 0; r < 8; ++r)
      sHid[wave][(r + 8*hi)*128 + j] = gelu_exact(c[r] + bias);
  }
  __syncthreads();
  if (lane < 16) {
    int pt = lane;
    float acc = fc2b[0];
    const float* hid = &sHid[wave][pt*128];
    for (int j = 0; j < 128; ++j) acc += hid[j] * fc2w[j];
    int p = pbase + pt;
    int nn = p >> 14, ppos = p & (PLANE-1);
    int y = ppos >> 7, x = ppos & 127;
    int bb = nn >> 4, t = nn & 15;
    out[(size_t)((bb*HH + y)*WW + x) * 16 + t] = acc;
  }
}

extern "C" void kernel_launch(void* const* d_in, const int* in_sizes, int n_in,
                              void* d_out, int out_size, void* d_ws, size_t ws_size,
                              hipStream_t stream) {
  (void)in_sizes; (void)n_in; (void)out_size; (void)ws_size;
  const float* xin  = (const float*)d_in[0];
  const float* fc0w = (const float*)d_in[1];
  const float* fc0b = (const float*)d_in[2];
  const float* w1r  = (const float*)d_in[3];
  const float* w1i  = (const float*)d_in[4];
  const float* w2r  = (const float*)d_in[5];
  const float* w2i  = (const float*)d_in[6];
  const float* skw  = (const float*)d_in[7];
  const float* skb  = (const float*)d_in[8];
  const float* fc1w = (const float*)d_in[9];
  const float* fc1b = (const float*)d_in[10];
  const float* fc2w = (const float*)d_in[11];
  const float* fc2b = (const float*)d_in[12];

  char* ws = (char*)d_ws;
  size_t off = 0;
  auto walloc = [&](size_t bytes) -> void* {
    void* p = ws + off;
    off = (off + bytes + 255) & ~(size_t)255;
    return p;
  };
  float* buf0 = (float*)walloc((size_t)NN*CC*PLANE*4);   // 134 MB
  float* buf1 = (float*)walloc((size_t)NN*CC*PLANE*4);   // 134 MB
  float* U    = (float*)walloc((size_t)NROWS*16*4);      // 16.8 MB
  float* Z    = (float*)walloc((size_t)NROWS*16*4);      // 16.8 MB
  float* V    = (float*)walloc((size_t)NN*CC*256*4);     // 2 MB
  float* Vo   = (float*)walloc((size_t)NN*CC*256*4);     // 2 MB
  _Float16* EwF  = (_Float16*)walloc(4*512*2);
  _Float16* EwiF = (_Float16*)walloc(8*512*2);
  _Float16* skF  = (_Float16*)walloc(4096*2);
  _Float16* fc1F = (_Float16*)walloc(4096*2);
  float* E1   = (float*)walloc(16*128*2*4);
  float* E1i  = (float*)walloc(128*16*2*4);

  fno_init_tables<<<16, 256, 0, stream>>>(EwF, EwiF, E1, E1i);
  fno_pack_weights<<<16, 256, 0, stream>>>(skw, fc1w, skF, fc1F);
  fno_fc0<<<NPTS/256, 256, 0, stream>>>(xin, fc0w, fc0b, buf0);

  float* cur = buf0;
  float* oth = buf1;
  for (int i = 0; i < 4; ++i) {
    fno_dft_w<<<NROWS/128, 256, 0, stream>>>(cur, U, EwF);
    fno_dft_h<<<(NN*CC*16*8)/256, 256, 0, stream>>>(U, V, E1);
    fno_specmul<<<(NN*16*8*32)/256, 256, 0, stream>>>(
        V, Vo, w1r + i*65536, w1i + i*65536, w2r + i*65536, w2i + i*65536);
    fno_idft_h<<<(NN*CC*HH*8)/256, 256, 0, stream>>>(Vo, Z, E1i);
    fno_idft_w<<<NROWS/128, 256, 0, stream>>>(Z, oth, EwiF);
    fno_skip<<<NPTS/128, 256, 0, stream>>>(cur, oth, skF + i*1024, skb + i*32, (i < 3) ? 1 : 0);
    float* tmp = cur; cur = oth; oth = tmp;
  }
  fno_fc<<<NPTS/128, 256, 0, stream>>>(cur, fc1F, fc1b, fc2w, fc2b, (float*)d_out);
}
